// Multihead_QMixer_25503515803774
// MI455X (gfx1250) — matrix operand order repaired and checked by reading, not compiled, not run on hardware
//
#include <hip/hip_runtime.h>
#include <hip/hip_bf16.h>

// ---------------------------------------------------------------------------
// Multihead QMixer for MI455X (gfx1250): all GEMMs have M = n_agents = 16, so
// every matmul maps onto v_wmma_f32_16x16x32_bf16 tiles. One block per batch
// element, 8 wave32s, activations in LDS (bf16), weights pre-converted to
// bf16 in workspace (L2-resident). A-fragments are hoisted into registers and
// reused across all N-tiles of each stage (they are N-invariant).
// ---------------------------------------------------------------------------

#define B_TOT 8192
#define A_N   16
#define OBS_D 128
#define S_D   320
#define E_D   160
#define H_N   5
#define HD_D  32
#define ED_D  64

typedef __attribute__((ext_vector_type(16))) __bf16 v16bf;
typedef __attribute__((ext_vector_type(8)))  float  v8f;

// LDS leading dims (bf16 elements), padded to avoid bank conflicts
#define LD_OBS 136   // 128 + 8
#define LD_E   168   // 160 + 8
#define LD_AT  40    // 32 + 8  (attn, zero padded K 16..31)
#define LD_H1  72    // 64 + 8
#define LD_HID 68    // f32 hidden, 64 + 4

// bf16 weight offsets inside workspace (elements)
#define WQ_OFF   0
#define WK_OFF   (WQ_OFF  + E_D*OBS_D)          // 20480
#define WV_OFF   (WK_OFF  + E_D*OBS_D)
#define WIN_OFF  (WV_OFF  + E_D*OBS_D)          // in_proj 480x160
#define WOUT_OFF (WIN_OFF + 3*E_D*E_D)
#define WH1_OFF  (WOUT_OFF+ E_D*E_D)            // 1024x160
#define WHB_OFF  (WH1_OFF + ED_D*A_N*E_D)
#define WHF_OFF  (WHB_OFF + ED_D*E_D)
#define WV1_OFF  (WHF_OFF + ED_D*E_D)
#define WS_ELEMS (WV1_OFF + ED_D*E_D)           // 358400 bf16 = 716800 B

// Native hardware converts (bf16 is a legal type on gfx1250 — feeds WMMA).
static __device__ __forceinline__ __bf16 f2bf(float f)  { return (__bf16)f; }
static __device__ __forceinline__ float  bf2f(__bf16 h) { return (float)h; }

// A fragment: 16x32 tile, row-major LDS matrix `base` with leading dim ld.
// Lane layout: row = lane&15, K = k0..k0+15 (16 contiguous K),
// k0 = 0 for lanes 0-15, 16 for lanes 16-31.
static __device__ __forceinline__ v16bf load_a(const __bf16* base, int ld,
                                               int kbase, int lane) {
    const int row = lane & 15;
    const int k0  = (lane < 16) ? 0 : 16;
    const __bf16* p = base + row * ld + kbase + k0;
    v16bf a;
#pragma unroll
    for (int i = 0; i < 16; ++i) a[i] = p[i];
    return a;
}

// B fragment from a weight stored [N][K] row-major (so B[k][n] = W[n][k]).
// Lane layout: col = lane&15, K = kb..kb+15 (kb = 0 / 16) -> 16 contiguous K.
static __device__ __forceinline__ v16bf load_b_wt(const __bf16* W, int ldk,
                                                  int nbase, int kbase, int lane) {
    const int col = lane & 15;
    const int kb  = (lane < 16) ? 0 : 16;
    const __bf16* p = W + (size_t)(nbase + col) * ldk + kbase + kb;
    v16bf b;
#pragma unroll
    for (int j = 0; j < 16; ++j) b[j] = p[j];
    return b;
}

// B fragment from a [K][N] row-major LDS matrix, with K clamp (rows >= kmax
// read as zero so zero-padded-A x garbage cannot produce NaN).
static __device__ __forceinline__ v16bf load_b_km(const __bf16* M, int ld,
                                                  int nbase, int kmax, int lane) {
    const int col = lane & 15;
    const int kb  = (lane < 16) ? 0 : 16;
    v16bf b;
#pragma unroll
    for (int j = 0; j < 16; ++j) {
        const int k = kb + j;
        b[j] = (k < kmax) ? M[(size_t)k * ld + nbase + col] : f2bf(0.f);
    }
    return b;
}

// Store C (f32 8xVGPR) + bias -> bf16 LDS tile. m = i + (lane<16?0:8).
// Bias pointer must be valid (no null check -> no branch in tile loops).
static __device__ __forceinline__ void store_c_bf16(v8f c, __bf16* dst, int ld,
                                                    int nbase,
                                                    const float* __restrict__ bias,
                                                    int lane) {
    const int col = lane & 15;
    const int mb  = (lane < 16) ? 0 : 8;
    const float bv = bias[nbase + col];
#pragma unroll
    for (int i = 0; i < 8; ++i)
        dst[(size_t)(mb + i) * ld + nbase + col] = f2bf(c[i] + bv);
}

static __device__ __forceinline__ void store_c_bf16_nb(v8f c, __bf16* dst, int ld,
                                                       int nbase, int lane) {
    const int col = lane & 15;
    const int mb  = (lane < 16) ? 0 : 8;
#pragma unroll
    for (int i = 0; i < 8; ++i)
        dst[(size_t)(mb + i) * ld + nbase + col] = f2bf(c[i]);
}

static __device__ __forceinline__ v8f wmma_bf16(v16bf a, v16bf b, v8f c) {
    return __builtin_amdgcn_wmma_f32_16x16x32_bf16(false, a, false, b,
                                                   (short)0, c, false, false);
}

// ---------------------------------------------------------------------------
__global__ void f32_to_bf16_kernel(const float* __restrict__ src,
                                   __bf16* __restrict__ dst, int n) {
    int i = blockIdx.x * blockDim.x + threadIdx.x;
    if (i < n) dst[i] = f2bf(src[i]);
}

// ---------------------------------------------------------------------------
__global__ __launch_bounds__(256)
void qmixer_kernel(const float* __restrict__ aq,     // (B,1,A)
                   const float* __restrict__ st,     // (B,1,S)
                   const float* __restrict__ obs,    // (B,A,OBS)
                   const float* __restrict__ qb, const float* __restrict__ kb_,
                   const float* __restrict__ vb, const float* __restrict__ inb,
                   const float* __restrict__ outb,
                   const float* __restrict__ hw1_b, const float* __restrict__ hb1_b,
                   const float* __restrict__ hwf_b,
                   const float* __restrict__ attW, const float* __restrict__ attb,
                   const float* __restrict__ v1b,
                   const float* __restrict__ v2W, const float* __restrict__ v2b,
                   const __bf16* __restrict__ ws,
                   float* __restrict__ out) {
    const int b    = blockIdx.x;
    const int tid  = threadIdx.x;
    const int wave = tid >> 5;
    const int lane = tid & 31;

    __shared__ __bf16 sObs[A_N * LD_OBS];
    __shared__ __bf16 sQv[A_N * LD_E], sKv[A_N * LD_E], sVv[A_N * LD_E];
    __shared__ __bf16 sQ [A_N * LD_E], sK [A_N * LD_E], sV [A_N * LD_E];
    __shared__ __bf16 sMh[A_N * LD_E];
    __shared__ __bf16 sMa[A_N * LD_E];
    __shared__ float  sScore[H_N * 16 * 16];
    __shared__ __bf16 sAttn[H_N * 16 * LD_AT];
    __shared__ float  sHid[A_N * LD_HID];
    __shared__ __bf16 sH1[A_N * LD_H1];
    __shared__ float  sY[A_N];
    __shared__ float  sAw[A_N];

    const __bf16* WQ   = ws + WQ_OFF;
    const __bf16* WK   = ws + WK_OFF;
    const __bf16* WV   = ws + WV_OFF;
    const __bf16* WIN  = ws + WIN_OFF;
    const __bf16* WOUT = ws + WOUT_OFF;
    const __bf16* WH1  = ws + WH1_OFF;
    const __bf16* WHB  = ws + WHB_OFF;
    const __bf16* WHF  = ws + WHF_OFF;
    const __bf16* WV1  = ws + WV1_OFF;

    // ---- Stage 0: obs tile (16x128 f32) -> bf16 LDS ----
    {
        const float* src = obs + (size_t)b * A_N * OBS_D;
        for (int i = tid; i < A_N * OBS_D; i += 256) {
            int r = i >> 7, c = i & 127;
            sObs[r * LD_OBS + c] = f2bf(src[i]);
        }
    }
    __syncthreads();

    // ---- Stage 1: qv/kv/vv = obs @ W.T + bias  (M16,K128,N160) ----
    // A (sObs) is shared by all 30 tiles: hoist 4 fragments into registers.
    {
        v16bf aO[4];
#pragma unroll
        for (int kt = 0; kt < 4; ++kt) aO[kt] = load_a(sObs, LD_OBS, kt * 32, lane);
#pragma unroll
        for (int mat = 0; mat < 3; ++mat) {
            const __bf16* W    = (mat == 0) ? WQ : (mat == 1) ? WK : WV;
            const float*  bias = (mat == 0) ? qb : (mat == 1) ? kb_ : vb;
            __bf16*       dst  = (mat == 0) ? sQv : (mat == 1) ? sKv : sVv;
            for (int nt = wave; nt < 10; nt += 8) {    // wave-uniform
                v8f c = {};
#pragma unroll
                for (int kt = 0; kt < 4; ++kt)
                    c = wmma_bf16(aO[kt], load_b_wt(W, OBS_D, nt * 16, kt * 32, lane), c);
                store_c_bf16(c, dst, LD_E, nt * 16, bias, lane);
            }
        }
    }
    __syncthreads();

    // ---- Stage 2: q/k/v = (qv|kv|vv) @ inW[slice].T + inb  (K160,N160) ----
#pragma unroll
    for (int mat = 0; mat < 3; ++mat) {
        const __bf16* src  = (mat == 0) ? sQv : (mat == 1) ? sKv : sVv;
        __bf16*       dst  = (mat == 0) ? sQ : (mat == 1) ? sK : sV;
        const __bf16* W    = WIN + (size_t)mat * E_D * E_D;
        const float*  bias = inb + mat * E_D;
        v16bf aS[5];
#pragma unroll
        for (int kt = 0; kt < 5; ++kt) aS[kt] = load_a(src, LD_E, kt * 32, lane);
        for (int nt = wave; nt < 10; nt += 8) {
            v8f c = {};
#pragma unroll
            for (int kt = 0; kt < 5; ++kt)
                c = wmma_bf16(aS[kt], load_b_wt(W, E_D, nt * 16, kt * 32, lane), c);
            store_c_bf16(c, dst, LD_E, nt * 16, bias, lane);
        }
    }
    __syncthreads();

    // ---- Stage 3: attention scores per head (16x16, K = head_dim = 32) ----
    for (int h = wave; h < H_N; h += 8) {
        v16bf a  = load_a(sQ, LD_E, h * HD_D, lane);
        v16bf bw = load_b_wt(sK, LD_E, 0, h * HD_D, lane);   // B[k][n]=K[n][k]
        v8f c = {};
        c = wmma_bf16(a, bw, c);
        const float scale = 0.17677669529663687f;            // 1/sqrt(32)
        const int col = lane & 15, mb = (lane < 16) ? 0 : 8;
#pragma unroll
        for (int i = 0; i < 8; ++i)
            sScore[h * 256 + (mb + i) * 16 + col] = c[i] * scale;
    }
    __syncthreads();

    // ---- softmax over k (VALU), write zero-padded bf16 attn tiles ----
    if (tid < H_N * 16) {
        const int h = tid >> 4, q = tid & 15;
        const float* row = sScore + h * 256 + q * 16;
        float m = row[0];
#pragma unroll
        for (int k = 1; k < 16; ++k) m = fmaxf(m, row[k]);
        float s = 0.f, e[16];
#pragma unroll
        for (int k = 0; k < 16; ++k) { e[k] = expf(row[k] - m); s += e[k]; }
        const float inv = 1.f / s;
        __bf16* drow = sAttn + h * 16 * LD_AT + q * LD_AT;
#pragma unroll
        for (int k = 0; k < 16; ++k) drow[k] = f2bf(e[k] * inv);
#pragma unroll
        for (int k = 16; k < 32; ++k) drow[k] = f2bf(0.f);   // zero K pad
    }
    __syncthreads();

    // ---- Stage 4: mh = attn @ vh  (K=16 zero-padded to 32, N=32/head) ----
    for (int job = wave; job < H_N * 2; job += 8) {
        const int h = job >> 1, dt = job & 1;
        v16bf a  = load_a(sAttn + h * 16 * LD_AT, LD_AT, 0, lane);
        v16bf bw = load_b_km(sV, LD_E, h * HD_D + dt * 16, 16, lane);
        v8f c = {};
        c = wmma_bf16(a, bw, c);
        store_c_bf16_nb(c, sMh, LD_E, h * HD_D + dt * 16, lane);
    }
    __syncthreads();

    // ---- Stage 5: ma_feats = mh @ outW.T + outb  (K160,N160) ----
    {
        v16bf aM[5];
#pragma unroll
        for (int kt = 0; kt < 5; ++kt) aM[kt] = load_a(sMh, LD_E, kt * 32, lane);
        for (int nt = wave; nt < 10; nt += 8) {
            v8f c = {};
#pragma unroll
            for (int kt = 0; kt < 5; ++kt)
                c = wmma_bf16(aM[kt], load_b_wt(WOUT, E_D, nt * 16, kt * 32, lane), c);
            store_c_bf16(c, sMa, LD_E, nt * 16, outb, lane);
        }
    }
    __syncthreads();

    // ---- ma_feats A-fragments: reused by stages 6, 7 and 9 ----
    v16bf aMa[5];
#pragma unroll
    for (int kt = 0; kt < 5; ++kt) aMa[kt] = load_a(sMa, LD_E, kt * 32, lane);

    // ---- Stage 6: hidden init = b1 = ma @ hb1_W.T + hb1_b  (N=64, f32) ----
    for (int nt = wave; nt < 4; nt += 8) {
        v8f c = {};
#pragma unroll
        for (int kt = 0; kt < 5; ++kt)
            c = wmma_bf16(aMa[kt], load_b_wt(WHB, E_D, nt * 16, kt * 32, lane), c);
        const int col = lane & 15, mb = (lane < 16) ? 0 : 8;
        const float bv = hb1_b[nt * 16 + col];
#pragma unroll
        for (int i = 0; i < 8; ++i)
            sHid[(mb + i) * LD_HID + nt * 16 + col] = c[i] + bv;
    }
    __syncthreads();

    // ---- Stage 7: hw1 (16x160 @ 160x1024) folded into Σ_n aq[n]*|w1| ----
    for (int t = wave; t < 64; t += 8) {          // 64 N-tiles of 16
        const int n_ag = t >> 2;                  // col = n_ag*64 + d
        const int dbase = (t & 3) * 16;
        v8f c = {};
#pragma unroll
        for (int kt = 0; kt < 5; ++kt)
            c = wmma_bf16(aMa[kt], load_b_wt(WH1, E_D, t * 16, kt * 32, lane), c);
        const int col = lane & 15, mb = (lane < 16) ? 0 : 8;
        const float bv  = hw1_b[t * 16 + col];
        const float aqn = aq[(size_t)b * A_N + n_ag];
#pragma unroll
        for (int i = 0; i < 8; ++i) {
            const float w1v = fabsf(c[i] + bv);
            atomicAdd(&sHid[(mb + i) * LD_HID + dbase + col], aqn * w1v);
        }
    }
    __syncthreads();

    // ---- Stage 8: elu(hidden); zero y accumulators ----
    for (int i = tid; i < A_N * ED_D; i += 256) {
        const int r = i >> 6, d = i & 63;
        const float x = sHid[r * LD_HID + d];
        sHid[r * LD_HID + d] = (x > 0.f) ? x : (expf(x) - 1.f);
    }
    if (tid < A_N) sY[tid] = 0.f;
    __syncthreads();

    // ---- Stage 9: wf reduction (waves 0-3) + v1 relu (waves 4-7) ----
    if (wave < 4) {
        const int nt = wave;
        v8f c = {};
#pragma unroll
        for (int kt = 0; kt < 5; ++kt)
            c = wmma_bf16(aMa[kt], load_b_wt(WHF, E_D, nt * 16, kt * 32, lane), c);
        const int col = lane & 15, mb = (lane < 16) ? 0 : 8;
        const float bv = hwf_b[nt * 16 + col];
#pragma unroll
        for (int i = 0; i < 8; ++i) {
            const float wfv = fabsf(c[i] + bv);
            atomicAdd(&sY[mb + i], wfv * sHid[(mb + i) * LD_HID + nt * 16 + col]);
        }
    } else {
        const int nt = wave - 4;
        v8f c = {};
#pragma unroll
        for (int kt = 0; kt < 5; ++kt)
            c = wmma_bf16(aMa[kt], load_b_wt(WV1, E_D, nt * 16, kt * 32, lane), c);
        const int col = lane & 15, mb = (lane < 16) ? 0 : 8;
        const float bv = v1b[nt * 16 + col];
#pragma unroll
        for (int i = 0; i < 8; ++i)
            sH1[(mb + i) * LD_H1 + nt * 16 + col] = f2bf(fmaxf(c[i] + bv, 0.f));
    }
    __syncthreads();

    // ---- Stage 10: vterm + attention-over-agents logits (16 lanes) ----
    if (tid < A_N) {
        float vt = v2b[0];
#pragma unroll 8
        for (int d = 0; d < ED_D; ++d) vt += bf2f(sH1[tid * LD_H1 + d]) * v2W[d];
        sY[tid] += vt;
        const float* srow = st + (size_t)b * S_D;
        const float* wrow = attW + (size_t)tid * S_D;
        float lg = attb[tid];
        for (int s = 0; s < S_D; ++s) lg += srow[s] * wrow[s];
        sAw[tid] = lg;
    }
    __syncthreads();

    // ---- Stage 11: softmax over agents, q_tot ----
    if (tid == 0) {
        float m = sAw[0];
#pragma unroll
        for (int a2 = 1; a2 < A_N; ++a2) m = fmaxf(m, sAw[a2]);
        float s = 0.f, acc = 0.f;
#pragma unroll
        for (int a2 = 0; a2 < A_N; ++a2) {
            const float e = expf(sAw[a2] - m);
            s += e;
            acc += e * sY[a2];
        }
        out[b] = acc / s;
    }
}

// ---------------------------------------------------------------------------
extern "C" void kernel_launch(void* const* d_in, const int* in_sizes, int n_in,
                              void* d_out, int out_size, void* d_ws, size_t ws_size,
                              hipStream_t stream) {
    (void)in_sizes; (void)n_in; (void)out_size; (void)ws_size;
    const float* aq    = (const float*)d_in[0];
    const float* st    = (const float*)d_in[1];
    const float* obs   = (const float*)d_in[2];
    const float* qW    = (const float*)d_in[3];
    const float* qb    = (const float*)d_in[4];
    const float* kW    = (const float*)d_in[5];
    const float* kb    = (const float*)d_in[6];
    const float* vW    = (const float*)d_in[7];
    const float* vb    = (const float*)d_in[8];
    const float* inW   = (const float*)d_in[9];
    const float* inb   = (const float*)d_in[10];
    const float* outW  = (const float*)d_in[11];
    const float* outb  = (const float*)d_in[12];
    const float* hw1_W = (const float*)d_in[13];
    const float* hw1_b = (const float*)d_in[14];
    const float* hb1_W = (const float*)d_in[15];
    const float* hb1_b = (const float*)d_in[16];
    const float* hwf_W = (const float*)d_in[17];
    const float* hwf_b = (const float*)d_in[18];
    const float* attW  = (const float*)d_in[19];
    const float* attb  = (const float*)d_in[20];
    const float* v1W   = (const float*)d_in[21];
    const float* v1b   = (const float*)d_in[22];
    const float* v2W   = (const float*)d_in[23];
    const float* v2b   = (const float*)d_in[24];

    __bf16* ws = (__bf16*)d_ws;

    // weight conversion prep (stream-ordered before the main kernel)
    struct Cvt { const float* src; int off; int n; };
    const Cvt cv[9] = {
        { qW,    WQ_OFF,   E_D * OBS_D },
        { kW,    WK_OFF,   E_D * OBS_D },
        { vW,    WV_OFF,   E_D * OBS_D },
        { inW,   WIN_OFF,  3 * E_D * E_D },
        { outW,  WOUT_OFF, E_D * E_D },
        { hw1_W, WH1_OFF,  ED_D * A_N * E_D },
        { hb1_W, WHB_OFF,  ED_D * E_D },
        { hwf_W, WHF_OFF,  ED_D * E_D },
        { v1W,   WV1_OFF,  ED_D * E_D },
    };
    for (int i = 0; i < 9; ++i) {
        const int n = cv[i].n;
        f32_to_bf16_kernel<<<(n + 255) / 256, 256, 0, stream>>>(
            cv[i].src, ws + cv[i].off, n);
    }

    qmixer_kernel<<<B_TOT, 256, 0, stream>>>(
        aq, st, obs, qb, kb, vb, inb, outb, hw1_b, hb1_b, hwf_b,
        attW, attb, v1b, v2W, v2b, ws, (float*)d_out);
}
